// GRU_47382079209745
// MI455X (gfx1250) — matrix operand order repaired and checked by reading, not compiled, not run on hardware
//
#include <hip/hip_runtime.h>
#include <hip/hip_bf16.h>

typedef __attribute__((ext_vector_type(16))) __bf16 v16bf;
typedef __attribute__((ext_vector_type(8)))  __bf16 v8bf;
typedef __attribute__((ext_vector_type(8)))  float  v8f;

#define DD   300
#define HH   1024
#define UU   64
#define LL   128
#define OO   7
#define SEQ  (UU*LL)   // 8192
#define KP   320       // D padded to multiple of 32
#define G3   (3*HH)    // 3072
#define NWG  64        // persistent workgroups in the recurrence

// ---------------- init: zero h ping-pong + sync words --------------------
__global__ void gru_init_kernel(float* hbuf, unsigned* sync) {
    int t = threadIdx.x;
    for (int i = t; i < 2 * HH; i += blockDim.x) hbuf[i] = 0.0f;
    if (t < 2) sync[t] = 0u;
}

// ---------------- convert W_ih (3072 x 300) -> bf16 padded (3072 x 320) --
__global__ void gru_cvt_wih_kernel(const float* __restrict__ W_ih,
                                   __bf16* __restrict__ Wb) {
    int idx = blockIdx.x * blockDim.x + threadIdx.x;
    if (idx >= G3 * KP) return;
    int row = idx / KP, k = idx % KP;
    Wb[idx] = (k < DD) ? (__bf16)W_ih[row * DD + k] : (__bf16)0.0f;
}

// ---------------- gather x = emb[tokens] -> bf16 padded (8192 x 320) -----
__global__ void gru_gather_kernel(const int* __restrict__ tokens,
                                  const float* __restrict__ emb,
                                  __bf16* __restrict__ xb) {
    int idx = blockIdx.x * blockDim.x + threadIdx.x;
    if (idx >= SEQ * KP) return;
    int row = idx / KP, k = idx % KP;
    float v = 0.0f;
    if (k < DD) {
        int tok = tokens[row];
        v = emb[(size_t)tok * DD + k];
    }
    xb[idx] = (__bf16)v;
}

// ---------------- gx = x @ W_ih.T + b_ih via WMMA bf16 -------------------
// Register-blocked: each wave computes a 32x64 tile of gx (2 M-subtiles x
// 4 N-subtiles = 8 accumulators), K loop over 320 in steps of 32.
// Per K-step: 6KB loads feed 8 WMMAs (0.75KB/WMMA vs 2KB unblocked).
__device__ __forceinline__ v16bf gru_load_a(const __bf16* arow, int kb, int half) {
    v8bf alo = *(const v8bf*)(arow + kb + half * 8);
    v8bf ahi = *(const v8bf*)(arow + kb + 16 + half * 8);
    v16bf a;
#pragma unroll
    for (int i = 0; i < 8; ++i) { a[i] = alo[i]; a[8 + i] = ahi[i]; }
    return a;
}

__global__ void __launch_bounds__(256)
gru_gemm_kernel(const __bf16* __restrict__ xb,
                const __bf16* __restrict__ Wb,
                const float*  __restrict__ b_ih,
                float* __restrict__ gx) {
    const int lane = threadIdx.x & 31;
    const int wave = threadIdx.x >> 5;
    const int tile = blockIdx.x * 8 + wave;          // 12288 waves total
    const int mT = tile / (G3 / 64);                 // 0..255 (32 rows each)
    const int nT = tile % (G3 / 64);                 // 0..47  (64 cols each)
    const int half = lane >> 4;                      // 0 or 1
    const int sub  = lane & 15;                      // m for A, n for B

    const __bf16* arow0 = xb + (size_t)(mT * 32 +      sub) * KP;
    const __bf16* arow1 = xb + (size_t)(mT * 32 + 16 + sub) * KP;
    const __bf16* brow0 = Wb + (size_t)(nT * 64 +      sub) * KP;
    const __bf16* brow1 = Wb + (size_t)(nT * 64 + 16 + sub) * KP;
    const __bf16* brow2 = Wb + (size_t)(nT * 64 + 32 + sub) * KP;
    const __bf16* brow3 = Wb + (size_t)(nT * 64 + 48 + sub) * KP;

    v8f c00 = {}, c01 = {}, c02 = {}, c03 = {};
    v8f c10 = {}, c11 = {}, c12 = {}, c13 = {};
#pragma unroll
    for (int kb = 0; kb < KP; kb += 32) {
        v16bf a0 = gru_load_a(arow0, kb, half);
        v16bf a1 = gru_load_a(arow1, kb, half);
        // B 32x16 bf16 (B[k][n] = W_ih[n][k]): same k order as A (8 at kb+8*half, 8 at kb+16+8*half)
        v16bf b0 = gru_load_a(brow0, kb, half);
        v16bf b1 = gru_load_a(brow1, kb, half);
        v16bf b2 = gru_load_a(brow2, kb, half);
        v16bf b3 = gru_load_a(brow3, kb, half);
        c00 = __builtin_amdgcn_wmma_f32_16x16x32_bf16(false, a0, false, b0, (short)0, c00, false, false);
        c01 = __builtin_amdgcn_wmma_f32_16x16x32_bf16(false, a0, false, b1, (short)0, c01, false, false);
        c02 = __builtin_amdgcn_wmma_f32_16x16x32_bf16(false, a0, false, b2, (short)0, c02, false, false);
        c03 = __builtin_amdgcn_wmma_f32_16x16x32_bf16(false, a0, false, b3, (short)0, c03, false, false);
        c10 = __builtin_amdgcn_wmma_f32_16x16x32_bf16(false, a1, false, b0, (short)0, c10, false, false);
        c11 = __builtin_amdgcn_wmma_f32_16x16x32_bf16(false, a1, false, b1, (short)0, c11, false, false);
        c12 = __builtin_amdgcn_wmma_f32_16x16x32_bf16(false, a1, false, b2, (short)0, c12, false, false);
        c13 = __builtin_amdgcn_wmma_f32_16x16x32_bf16(false, a1, false, b3, (short)0, c13, false, false);
    }
    // C layout: VGPR r holds row (8*half + r) of the 16x16 subtile, col = sub
    v8f acc[2][4] = {{c00, c01, c02, c03}, {c10, c11, c12, c13}};
#pragma unroll
    for (int q = 0; q < 4; ++q) {
        const int col  = nT * 64 + q * 16 + sub;
        const float bias = b_ih[col];
#pragma unroll
        for (int mi = 0; mi < 2; ++mi) {
            const int mBase = mT * 32 + mi * 16 + 8 * half;
#pragma unroll
            for (int r = 0; r < 8; ++r)
                gx[(size_t)(mBase + r) * G3 + col] = acc[mi][q][r] + bias;
        }
    }
}

// ---------------- persistent recurrence --------------------------------
__device__ __forceinline__ void grid_barrier(unsigned* cnt, unsigned* gen,
                                             unsigned t) {
    __syncthreads();
    if (threadIdx.x == 0) {
        unsigned prev = __hip_atomic_fetch_add(cnt, 1u, __ATOMIC_ACQ_REL,
                                               __HIP_MEMORY_SCOPE_AGENT);
        if (prev == NWG * (t + 1u) - 1u) {
            __hip_atomic_store(gen, t + 1u, __ATOMIC_RELEASE,
                               __HIP_MEMORY_SCOPE_AGENT);
        } else {
            while (__hip_atomic_load(gen, __ATOMIC_ACQUIRE,
                                     __HIP_MEMORY_SCOPE_AGENT) < t + 1u) {
                asm volatile("s_sleep 1");
            }
        }
    }
    __syncthreads();
}

__device__ __forceinline__ float sigmoidf_(float x) {
    return 1.0f / (1.0f + expf(-x));
}

__global__ void __launch_bounds__(512, 1)
gru_recur_kernel(const float* __restrict__ gx,
                 const float* __restrict__ W_hh,
                 const float* __restrict__ b_hh,
                 float* __restrict__ hbuf,     // [2][H] ping-pong
                 float* __restrict__ hlast,    // [U][H]
                 unsigned* __restrict__ sync) {
    __shared__ float hsh[HH];
    const int wave = threadIdx.x >> 5;           // 0..15
    const int lane = threadIdx.x & 31;
    const int j = blockIdx.x * 16 + wave;        // 0..1023 output index

    const float* wr = W_hh + (size_t)j * HH;
    const float* wz = W_hh + (size_t)(HH + j) * HH;
    const float* wn = W_hh + (size_t)(2 * HH + j) * HH;
    const float bhr = b_hh[j], bhz = b_hh[HH + j], bhn = b_hh[2 * HH + j];

    for (unsigned t = 0; t < SEQ; ++t) {
        grid_barrier(sync, sync + 1, t);         // step t-1 fully visible
        // stage h_prev into LDS (1024 floats, 512 threads)
        const float* hp = hbuf + (size_t)(t & 1u) * HH;
        hsh[threadIdx.x]       = hp[threadIdx.x];
        hsh[threadIdx.x + 512] = hp[threadIdx.x + 512];
        __syncthreads();

        float ar = 0.f, az = 0.f, an = 0.f;
#pragma unroll 4
        for (int k = lane; k < HH; k += 32) {
            float hv = hsh[k];
            ar = fmaf(wr[k], hv, ar);
            az = fmaf(wz[k], hv, az);
            an = fmaf(wn[k], hv, an);
        }
#pragma unroll
        for (int off = 16; off > 0; off >>= 1) {
            ar += __shfl_xor(ar, off, 32);
            az += __shfl_xor(az, off, 32);
            an += __shfl_xor(an, off, 32);
        }
        if (lane == 0) {
            const float* g = gx + (size_t)t * G3;
            float r  = sigmoidf_(g[j]          + ar + bhr);
            float z  = sigmoidf_(g[HH + j]     + az + bhz);
            float n  = tanhf(g[2 * HH + j] + r * (an + bhn));
            float hn = (1.0f - z) * n + z * hsh[j];
            hbuf[(size_t)((t + 1u) & 1u) * HH + j] = hn;
            if ((t & (LL - 1)) == (LL - 1))
                hlast[(size_t)(t >> 7) * HH + j] = hn;
        }
    }
}

// ---------------- out = h_last @ W_out.T + b_out -------------------------
__global__ void gru_out_kernel(const float* __restrict__ hlast,
                               const float* __restrict__ W_out,
                               const float* __restrict__ b_out,
                               float* __restrict__ out) {
    const int u = blockIdx.x;                     // 0..63
    const int o = threadIdx.x >> 5;               // 0..6 (7 waves)
    const int lane = threadIdx.x & 31;
    const float* h = hlast + (size_t)u * HH;
    const float* w = W_out + (size_t)o * HH;
    float s = 0.f;
    for (int k = lane; k < HH; k += 32) s = fmaf(h[k], w[k], s);
#pragma unroll
    for (int off = 16; off > 0; off >>= 1) s += __shfl_xor(s, off, 32);
    if (lane == 0) out[u * OO + o] = s + b_out[o];
}

// ------------------------------------------------------------------------
static inline size_t align256(size_t x) { return (x + 255) & ~(size_t)255; }

extern "C" void kernel_launch(void* const* d_in, const int* in_sizes, int n_in,
                              void* d_out, int out_size, void* d_ws, size_t ws_size,
                              hipStream_t stream) {
    const int*   tokens = (const int*)  d_in[0];
    const float* emb    = (const float*)d_in[1];
    const float* W_ih   = (const float*)d_in[2];
    const float* W_hh   = (const float*)d_in[3];
    const float* b_ih   = (const float*)d_in[4];
    const float* b_hh   = (const float*)d_in[5];
    const float* W_out  = (const float*)d_in[6];
    const float* b_out  = (const float*)d_in[7];
    float* out = (float*)d_out;

    char* ws = (char*)d_ws;
    size_t off = 0;
    __bf16* xb = (__bf16*)(ws + off); off = align256(off + (size_t)SEQ * KP * sizeof(__bf16));
    __bf16* Wb = (__bf16*)(ws + off); off = align256(off + (size_t)G3  * KP * sizeof(__bf16));
    float*  gx = (float*) (ws + off); off = align256(off + (size_t)SEQ * G3 * sizeof(float));
    float*  hbuf  = (float*)(ws + off); off = align256(off + (size_t)2 * HH * sizeof(float));
    float*  hlast = (float*)(ws + off); off = align256(off + (size_t)UU * HH * sizeof(float));
    unsigned* sync = (unsigned*)(ws + off);

    gru_init_kernel<<<1, 256, 0, stream>>>(hbuf, sync);
    gru_cvt_wih_kernel<<<(G3 * KP + 255) / 256, 256, 0, stream>>>(W_ih, Wb);
    gru_gather_kernel<<<(SEQ * KP + 255) / 256, 256, 0, stream>>>(tokens, emb, xb);

    // 256 M-blocks x 48 N-blocks = 12288 waves, 8 waves per 256-thread block
    gru_gemm_kernel<<<(SEQ / 32) * (G3 / 64) / 8, 256, 0, stream>>>(xb, Wb, b_ih, gx);

    gru_recur_kernel<<<NWG, 512, 0, stream>>>(gx, W_hh, b_hh, hbuf, hlast, sync);

    gru_out_kernel<<<UU, 32 * OO, 0, stream>>>(hlast, W_out, b_out, out);
}
